// TwinBottleneck_20212116095716
// MI455X (gfx1250) — matrix operand order repaired and checked by reading, not compiled, not run on hardware
//
#include <hip/hip_runtime.h>
#include <hip/hip_bf16.h>
#include <stdint.h>

// ---------------------------------------------------------------------------
// TwinBottleneck GCN block for MI455X (gfx1250, wave32, WMMA).
//   out = sigmoid( D^-1/2 A D^-1/2 (cbn @ W + b) )
//   A_ij = (1 - |2*(bbn_i . bbn_j) - s_i - s_j| / 32)^1.4   (exactly symmetric)
// All three matmuls run on v_wmma_f32_16x16x32_bf16 (8192 MAC/instr).
// Elementwise transcendentals use raw v_log_f32/v_exp_f32/v_rcp_f32 builtins
// ---------------------------------------------------------------------------

typedef __attribute__((ext_vector_type(16))) __bf16 v16bf;
typedef __attribute__((ext_vector_type(8)))  __bf16 v8bf;
typedef __attribute__((ext_vector_type(8)))  float  v8f;

#define NN   8192
#define BBN  32
#define CBN  512

#define LOG2E 1.4426950408889634f

// pow(x, 1.4) = exp2(1.4 * log2(x)); x<0 -> NaN (matches jnp.power), 0 -> 0.
static __device__ __forceinline__ float fast_pow14(float x) {
  return __builtin_amdgcn_exp2f(1.4f * __builtin_amdgcn_logf(x));
}

// sigmoid(v) = rcp(1 + 2^(-v*log2e)) : v_exp_f32 + v_rcp_f32
static __device__ __forceinline__ float fast_sigmoid(float v) {
  return __builtin_amdgcn_rcpf(1.f + __builtin_amdgcn_exp2f(-v * LOG2E));
}

// Build a 16-element bf16 fragment from two aligned 16-byte chunks.
static __device__ __forceinline__ v16bf frag2x16B(const __bf16* p0, const __bf16* p1) {
  v8bf lo = *(const v8bf*)p0;
  v8bf hi = *(const v8bf*)p1;
  v16bf r;
#pragma unroll
  for (int i = 0; i < 8; ++i) { r[i] = lo[i]; r[i + 8] = hi[i]; }
  return r;
}

static __device__ __forceinline__ v8f wmma_bf16(v16bf a, v16bf b, v8f c) {
  return __builtin_amdgcn_wmma_f32_16x16x32_bf16(false, a, false, b, (short)0, c,
                                                 false, false);
}

// --- prep: bbn -> bf16, s = rowsum(bbn), d = 0 ------------------------------
__global__ void k_prep_bbn(const float* __restrict__ bbn, __bf16* __restrict__ bbn_bf,
                           float* __restrict__ s, float* __restrict__ d) {
  int row = blockIdx.x * blockDim.x + threadIdx.x;
  if (row >= NN) return;
  const float* src = bbn + row * BBN;
  float sum = 0.f;
#pragma unroll
  for (int k = 0; k < BBN; ++k) {
    float v = src[k];
    sum += v;
    bbn_bf[row * BBN + k] = (__bf16)v;
  }
  s[row] = sum;
  d[row] = 0.f;  // degree accumulator, re-zeroed every launch (graph-safe)
}

__global__ void k_cvt_bf16(const float* __restrict__ src, __bf16* __restrict__ dst, int n) {
  int i = blockIdx.x * blockDim.x + threadIdx.x;
  if (i < n) dst[i] = (__bf16)src[i];
}

// Wt[c][k] = W[k][c]  (so WMMA B-fragments read contiguous K-pairs)
__global__ void k_prep_w(const float* __restrict__ W, __bf16* __restrict__ Wt) {
  int idx = blockIdx.x * blockDim.x + threadIdx.x;  // 512*512
  int c = idx >> 9, k = idx & 511;
  Wt[idx] = (__bf16)W[k * CBN + c];
}

// --- adjacency: one wmma per 16x16 Gram tile (K = 32 exactly) ---------------
// Block: 256 thr (8 waves), 16 rows (i0) per block, blockIdx.y picks a 2048-col
// chunk; each wave sweeps 16 j-tiles. Stores bf16 adj (symmetric: stored [j][i])
// and accumulates degree row-sums via shfl_xor + one atomicAdd per row-chunk.
__global__ void __launch_bounds__(256) k_adj(const __bf16* __restrict__ bbn_bf,
                                             const float* __restrict__ s,
                                             __bf16* __restrict__ adj,
                                             float* __restrict__ d) {
  const int lane = threadIdx.x & 31;
  const int wave = threadIdx.x >> 5;
  const int hi16 = (lane >= 16) ? 1 : 0;
  const int lm = lane & 15;
  const int i0 = blockIdx.x * 16;

  // A fragment (16x32 bf16): lane row = i0+lm; K chunks per ISA layout.
  const __bf16* arow = bbn_bf + (size_t)(i0 + lm) * BBN;
  const int aoff = hi16 ? 8 : 0;
  const v16bf afrag = frag2x16B(arow + aoff, arow + aoff + 16);

  float si[8];
#pragma unroll
  for (int r = 0; r < 8; ++r) si[r] = s[i0 + r + hi16 * 8];

  float dacc[8];
#pragma unroll
  for (int r = 0; r < 8; ++r) dacc[r] = 0.f;

  const int jbase = blockIdx.y * 2048;
  const int boff = hi16 ? 8 : 0;
  for (int t = 0; t < 16; ++t) {
    const int j0 = jbase + ((wave + (t << 3)) << 4);
    // B fragment (32x16): B[k][n] = bbn[j0+n][k] -> two 16-byte chunks per lane, same k order as A.
    const __bf16* brow = bbn_bf + (size_t)(j0 + lm) * BBN;
    const v16bf bfrag = frag2x16B(brow + boff, brow + boff + 16);
    v8f g = {};
    g = wmma_bf16(afrag, bfrag, g);
    const float sj = s[j0 + lm];
#pragma unroll
    for (int p = 0; p < 4; ++p) {
      float nd0 = fabsf(2.f * g[2 * p]     - si[2 * p]     - sj) * (1.f / 32.f);
      float nd1 = fabsf(2.f * g[2 * p + 1] - si[2 * p + 1] - sj) * (1.f / 32.f);
      float a0 = fast_pow14(1.f - nd0);
      float a1 = fast_pow14(1.f - nd1);
      dacc[2 * p] += a0;
      dacc[2 * p + 1] += a1;
      union { __bf16 b[2]; uint32_t u; } pk;
      pk.b[0] = (__bf16)a0;
      pk.b[1] = (__bf16)a1;
      const int m0 = 2 * p + hi16 * 8;  // adjacent rows -> contiguous in adj^T
      *(uint32_t*)(adj + (size_t)(j0 + lm) * NN + i0 + m0) = pk.u;  // symmetric
    }
  }
#pragma unroll
  for (int r = 0; r < 8; ++r) {
    float v = dacc[r];
    v += __shfl_xor(v, 1, 32);
    v += __shfl_xor(v, 2, 32);
    v += __shfl_xor(v, 4, 32);
    v += __shfl_xor(v, 8, 32);
    if (lm == 0) atomicAdd(&d[i0 + r + hi16 * 8], v);
  }
}

__global__ void k_dinv(const float* __restrict__ d, float* __restrict__ dinv) {
  int i = blockIdx.x * blockDim.x + threadIdx.x;
  if (i < NN) dinv[i] = __builtin_amdgcn_rsqf(d[i] + 1e-8f);
}

// --- fc: fcT[c][j] = bf16( dinv_j * (cbn[j] @ W[:,c] + b[c]) ) --------------
// Block: 16 j-rows x 512 c-cols; wave w owns 16x64 (4 C tiles, 32 acc VGPRs).
__global__ void __launch_bounds__(256) k_fc(const __bf16* __restrict__ cbn_bf,
                                            const __bf16* __restrict__ Wt,
                                            const float* __restrict__ bias,
                                            const float* __restrict__ dinv,
                                            __bf16* __restrict__ fcT) {
  const int lane = threadIdx.x & 31;
  const int wave = threadIdx.x >> 5;
  const int hi16 = (lane >= 16) ? 1 : 0;
  const int lm = lane & 15;
  const int j0 = blockIdx.x * 16;
  const int aoff = hi16 ? 8 : 0;
  const int boff = hi16 ? 8 : 0;

  v8f acc[4];
#pragma unroll
  for (int t = 0; t < 4; ++t) acc[t] = (v8f){};

  const __bf16* arowb = cbn_bf + (size_t)(j0 + lm) * CBN;
  for (int k0 = 0; k0 < CBN; k0 += 32) {
    const v16bf a = frag2x16B(arowb + k0 + aoff, arowb + k0 + aoff + 16);
#pragma unroll
    for (int t = 0; t < 4; ++t) {
      const int c0 = wave * 64 + t * 16;
      const __bf16* brow = Wt + (size_t)(c0 + lm) * CBN + k0;
      const v16bf b = frag2x16B(brow + boff, brow + boff + 16);
      acc[t] = wmma_bf16(a, b, acc[t]);
    }
  }

  float di[8];
#pragma unroll
  for (int r = 0; r < 8; ++r) di[r] = dinv[j0 + r + hi16 * 8];

#pragma unroll
  for (int t = 0; t < 4; ++t) {
    const int c = wave * 64 + t * 16 + lm;
    const float bc = bias[c];
#pragma unroll
    for (int p = 0; p < 4; ++p) {
      const int m0 = 2 * p + hi16 * 8;
      float v0 = (acc[t][2 * p]     + bc) * di[2 * p];
      float v1 = (acc[t][2 * p + 1] + bc) * di[2 * p + 1];
      union { __bf16 b[2]; uint32_t u; } pk;
      pk.b[0] = (__bf16)v0;
      pk.b[1] = (__bf16)v1;
      *(uint32_t*)(fcT + (size_t)c * NN + j0 + m0) = pk.u;  // transposed store
    }
  }
}

// --- conv: out = sigmoid( dinv_i * (adj @ fcT^T) ) -- the 34.4 GMAC GEMM ----
__global__ void __launch_bounds__(256) k_conv(const __bf16* __restrict__ adj,
                                              const __bf16* __restrict__ fcT,
                                              const float* __restrict__ dinv,
                                              float* __restrict__ out) {
  const int lane = threadIdx.x & 31;
  const int wave = threadIdx.x >> 5;
  const int hi16 = (lane >= 16) ? 1 : 0;
  const int lm = lane & 15;
  const int i0 = blockIdx.x * 16;
  const int aoff = hi16 ? 8 : 0;
  const int boff = hi16 ? 8 : 0;

  v8f acc[4];
#pragma unroll
  for (int t = 0; t < 4; ++t) acc[t] = (v8f){};

  const __bf16* arowb = adj + (size_t)(i0 + lm) * NN;
  for (int k0 = 0; k0 < NN; k0 += 32) {
    const v16bf a = frag2x16B(arowb + k0 + aoff, arowb + k0 + aoff + 16);
#pragma unroll
    for (int t = 0; t < 4; ++t) {
      const int c0 = wave * 64 + t * 16;
      const __bf16* brow = fcT + (size_t)(c0 + lm) * NN + k0;
      const v16bf b = frag2x16B(brow + boff, brow + boff + 16);
      acc[t] = wmma_bf16(a, b, acc[t]);
    }
  }

  float di[8];
#pragma unroll
  for (int r = 0; r < 8; ++r) di[r] = dinv[i0 + r + hi16 * 8];

#pragma unroll
  for (int t = 0; t < 4; ++t) {
    const int c = wave * 64 + t * 16 + lm;
#pragma unroll
    for (int r = 0; r < 8; ++r) {
      const int m = r + hi16 * 8;
      out[(size_t)(i0 + m) * CBN + c] = fast_sigmoid(acc[t][r] * di[r]);
    }
  }
}

// ---------------------------------------------------------------------------
extern "C" void kernel_launch(void* const* d_in, const int* in_sizes, int n_in,
                              void* d_out, int out_size, void* d_ws, size_t ws_size,
                              hipStream_t stream) {
  const float* bbn = (const float*)d_in[0];
  const float* cbn = (const float*)d_in[1];
  const float* W   = (const float*)d_in[2];
  const float* b   = (const float*)d_in[3];
  float* out = (float*)d_out;

  // Workspace carve-out (~145 MB; adj bf16 + fc bf16 are L2-resident at 192MB).
  char* ws = (char*)d_ws;
  size_t off = 0;
  auto carve = [&](size_t bytes) -> char* {
    char* p = ws + off;
    off = (off + bytes + 255) & ~(size_t)255;
    return p;
  };
  __bf16* bbn_bf = (__bf16*)carve((size_t)NN * BBN * 2);
  __bf16* cbn_bf = (__bf16*)carve((size_t)NN * CBN * 2);
  __bf16* Wt     = (__bf16*)carve((size_t)CBN * CBN * 2);
  __bf16* fcT    = (__bf16*)carve((size_t)CBN * NN * 2);
  float*  s      = (float*)carve(NN * 4);
  float*  d      = (float*)carve(NN * 4);
  float*  dinv   = (float*)carve(NN * 4);
  __bf16* adj    = (__bf16*)carve((size_t)NN * NN * 2);
  (void)ws_size; (void)in_sizes; (void)n_in; (void)out_size;

  k_prep_bbn<<<NN / 256, 256, 0, stream>>>(bbn, bbn_bf, s, d);
  k_cvt_bf16<<<(NN * CBN) / 256, 256, 0, stream>>>(cbn, cbn_bf, NN * CBN);
  k_prep_w<<<(CBN * CBN) / 256, 256, 0, stream>>>(W, Wt);
  k_adj<<<dim3(NN / 16, 4), 256, 0, stream>>>(bbn_bf, s, adj, d);
  k_dinv<<<NN / 256, 256, 0, stream>>>(d, dinv);
  k_fc<<<NN / 16, 256, 0, stream>>>(cbn_bf, Wt, b, dinv, fcT);
  k_conv<<<NN / 16, 256, 0, stream>>>(adj, fcT, dinv, out);
}
